// Qwen3FlashMTPAttention_45122926412363
// MI455X (gfx1250) — matrix operand order repaired and checked by reading, not compiled, not run on hardware
//
#include <hip/hip_runtime.h>
#include <hip/hip_bf16.h>
#include <math.h>
#include <stdint.h>

// ---------------------------------------------------------------------------
// Qwen3 attention block for MI455X (gfx1250, wave32, WMMA 16x16x32 f16/f32acc)
//
// Pipeline:
//   1) cvt X -> f16; transpose+cvt Wq/Wk/Wv/Wo -> f16 [N][K]
//   2) WMMA GEMM: Qf/Kf/Vf (f32) = X @ W           (async global->LDS staging)
//   3) normrope: RMSnorm(q,k) + RoPE + 1/sqrt(128) folded into Q
//   4) flash attention (online softmax, causal), WMMA for QK^T and PV
//   5) WMMA GEMM: out = attn @ Wo (f32 to d_out)
// ---------------------------------------------------------------------------

typedef _Float16 h16x16 __attribute__((ext_vector_type(16)));
typedef _Float16 h16x8  __attribute__((ext_vector_type(8)));
typedef float    f32x8  __attribute__((ext_vector_type(8)));

#define HD     128
#define NH     16
#define NKV    8
#define BATCH  2
#define SEQ    2048
#define HID    2048
#define BS     (BATCH * SEQ)
#define QSC    0.08838834764831845f   // 1/sqrt(128)

// ---------------------------------------------------------------------------
// CDNA5 async global->LDS copy (ASYNCcnt-tracked), guarded so either toolchain
// compiles. Fallback = global->VGPR->ds_store.
// Builtin signature (probe-confirmed by diagnostics): param0 = v4i in AS1
// (printed '__device__'), so we cast to int4 pointers with explicit AS.
// ---------------------------------------------------------------------------
#if defined(__gfx1250__) && \
    __has_builtin(__builtin_amdgcn_global_load_async_to_lds_b128) && \
    __has_builtin(__builtin_amdgcn_s_wait_asynccnt)
#define HAVE_ASYNC_COPY 1
#else
#define HAVE_ASYNC_COPY 0
#endif

#if HAVE_ASYNC_COPY
typedef int v4i __attribute__((ext_vector_type(4)));
typedef __attribute__((address_space(1))) v4i as1_v4i;
typedef __attribute__((address_space(3))) v4i as3_v4i;
__device__ __forceinline__ void async_copy16(const void* g, void* l) {
    // generic->AS1 keeps the 64-bit VA; generic LDS VA low 32 bits are the
    // LDS offset.
    __builtin_amdgcn_global_load_async_to_lds_b128(
        (as1_v4i*)(uintptr_t)g, (as3_v4i*)(uint32_t)(uintptr_t)l, 0, 0);
}
__device__ __forceinline__ void async_wait0() {
    __builtin_amdgcn_s_wait_asynccnt(0);
}
#endif

// ---------------------------------------------------------------------------
// elementwise converts
// ---------------------------------------------------------------------------
__global__ void cvt_f32_to_h16(const float* __restrict__ in,
                               _Float16* __restrict__ out, size_t n) {
    size_t i = (size_t)blockIdx.x * blockDim.x + threadIdx.x;
    if (i < n) out[i] = (_Float16)in[i];
}

// W [K][N] row-major f32  ->  Wt [N][K] row-major f16
__global__ void transpose_to_h16(const float* __restrict__ W,
                                 _Float16* __restrict__ Wt, int K, int N) {
    size_t i = (size_t)blockIdx.x * blockDim.x + threadIdx.x;
    if (i < (size_t)K * N) {
        int k = (int)(i / N), n = (int)(i % N);
        Wt[(size_t)n * K + k] = (_Float16)W[i];
    }
}

// ---------------------------------------------------------------------------
// WMMA GEMM: C[M][N] f32 = A[M][K] f16 @ Bt[N][K]^T f16
// block = 256 thr = 8 waves; tile 128(M) x 64(N); k-step 32
// wave grid 4x2 -> each wave computes 32x32 via 2x2 WMMA 16x16x32 tiles
// ---------------------------------------------------------------------------
#define TM 128
#define TN 64
#define TK 32

__global__ __launch_bounds__(256) void gemm_h16_f32(
        const _Float16* __restrict__ A, const _Float16* __restrict__ Bt,
        float* __restrict__ C, int M, int N, int K) {
    __shared__ _Float16 As[TM][TK + 8];   // row stride 80B (16B aligned)
    __shared__ _Float16 Bs[TN][TK + 8];

    const int tid  = threadIdx.x;
    const int lane = tid & 31, l16 = lane & 15, hi = lane >> 4;
    const int w    = tid >> 5;
    const int wm   = w >> 1, wn = w & 1;
    const int m0   = blockIdx.y * TM;
    const int n0   = blockIdx.x * TN;

    const int ra = tid >> 1, ca = (tid & 1) * 16;   // A staging slot
    const int rb = tid >> 2, cb = (tid & 3) * 8;    // B staging slot

    f32x8 acc[2][2] = {};

    for (int k0 = 0; k0 < K; k0 += TK) {
        const _Float16* ga = A  + (size_t)(m0 + ra) * K + k0 + ca;
        const _Float16* gb = Bt + (size_t)(n0 + rb) * K + k0 + cb;
#if HAVE_ASYNC_COPY
        // direct HBM/L2 -> LDS, tracked by ASYNCcnt (no VGPR round trip)
        async_copy16(ga,     &As[ra][ca]);
        async_copy16(ga + 8, &As[ra][ca + 8]);
        async_copy16(gb,     &Bs[rb][cb]);
        if (k0 + TK < K) {                 // pull next k-tile toward L2/L0
            __builtin_prefetch(ga + TK, 0, 1);
            __builtin_prefetch(gb + TK, 0, 1);
        }
        async_wait0();
#else
        {
            const h16x8* g = (const h16x8*)ga;
            *(h16x8*)&As[ra][ca]     = g[0];
            *(h16x8*)&As[ra][ca + 8] = g[1];
            *(h16x8*)&Bs[rb][cb]     = *(const h16x8*)gb;
        }
#endif
        __syncthreads();

        h16x16 afrag[2], bfrag[2];
        #pragma unroll
        for (int mi = 0; mi < 2; ++mi) {
            // A 16x32 layout: lanes0-15 K=0..7,16..23 ; lanes16-31 K=8..15,24..31
            const _Float16* ap = &As[wm * 32 + mi * 16 + l16][hi * 8];
            #pragma unroll
            for (int i = 0; i < 8; ++i) { afrag[mi][i] = ap[i]; afrag[mi][8 + i] = ap[16 + i]; }
        }
        #pragma unroll
        for (int ni = 0; ni < 2; ++ni) {
            // B 32x16 layout: lane = column; lanes0-15 K=0..7,16..23 ; lanes16-31 K=8..15,24..31
            const _Float16* bp = &Bs[wn * 32 + ni * 16 + l16][hi * 8];
            #pragma unroll
            for (int i = 0; i < 8; ++i) { bfrag[ni][i] = bp[i]; bfrag[ni][8 + i] = bp[16 + i]; }
        }
        #pragma unroll
        for (int mi = 0; mi < 2; ++mi)
            #pragma unroll
            for (int ni = 0; ni < 2; ++ni)
                acc[mi][ni] = __builtin_amdgcn_wmma_f32_16x16x32_f16(
                    false, afrag[mi], false, bfrag[ni], (short)0,
                    acc[mi][ni], false, false);
        __syncthreads();
    }

    // C layout: VGPR v -> rows v (lanes0-15) / v+8 (lanes16-31), col = l16
    #pragma unroll
    for (int mi = 0; mi < 2; ++mi)
        #pragma unroll
        for (int ni = 0; ni < 2; ++ni) {
            int row = m0 + wm * 32 + mi * 16 + hi * 8;
            int col = n0 + wn * 32 + ni * 16 + l16;
            float* cp = C + (size_t)row * N + col;
            #pragma unroll
            for (int v = 0; v < 8; ++v) cp[(size_t)v * N] = acc[mi][ni][v];
        }
}

// ---------------------------------------------------------------------------
// RMS-norm + RoPE + layout change. grid (BS, 32), block 128 (= head_dim)
// slot 0..15 -> Q head, 16..23 -> K head, 24..31 -> V head
// ---------------------------------------------------------------------------
__global__ __launch_bounds__(128) void normrope(
        const float* __restrict__ Qf, const float* __restrict__ Kf,
        const float* __restrict__ Vf, const float* __restrict__ cosb,
        const float* __restrict__ sinb, const float* __restrict__ qw,
        const float* __restrict__ kw, _Float16* __restrict__ Qh,
        _Float16* __restrict__ Kh, _Float16* __restrict__ Vh) {
    const int tok  = blockIdx.x;
    const int slot = blockIdx.y;
    const int d    = threadIdx.x;
    const int b = tok / SEQ, s = tok % SEQ;
    __shared__ float red[4];

    const float c  = cosb[(size_t)tok * HD + d];
    const float sn = sinb[(size_t)tok * HD + d];
    const int   dp = (d < 64) ? d + 64 : d - 64;

    if (slot < NH) {                    // Q: rmsnorm + rope + score scaling
        int h = slot;
        const float* row = Qf + (size_t)tok * (NH * HD) + (size_t)h * HD;
        float x = row[d], xp = row[dp];
        float ss = x * x;
        for (int off = 16; off; off >>= 1) ss += __shfl_xor(ss, off, 32);
        if ((threadIdx.x & 31) == 0) red[threadIdx.x >> 5] = ss;
        __syncthreads();
        ss = red[0] + red[1] + red[2] + red[3];
        float inv = rsqrtf(ss * (1.0f / HD) + 1e-6f);
        float xn  = x * inv * qw[d];
        float xpn = xp * inv * qw[dp];
        float rot = (d < 64) ? -xpn : xpn;
        float out = (xn * c + rot * sn) * QSC;
        Qh[((size_t)(b * NH + h) * SEQ + s) * HD + d] = (_Float16)out;
    } else if (slot < NH + NKV) {       // K: rmsnorm + rope
        int h = slot - NH;
        const float* row = Kf + (size_t)tok * (NKV * HD) + (size_t)h * HD;
        float x = row[d], xp = row[dp];
        float ss = x * x;
        for (int off = 16; off; off >>= 1) ss += __shfl_xor(ss, off, 32);
        if ((threadIdx.x & 31) == 0) red[threadIdx.x >> 5] = ss;
        __syncthreads();
        ss = red[0] + red[1] + red[2] + red[3];
        float inv = rsqrtf(ss * (1.0f / HD) + 1e-6f);
        float xn  = x * inv * kw[d];
        float xpn = xp * inv * kw[dp];
        float rot = (d < 64) ? -xpn : xpn;
        float out = xn * c + rot * sn;
        Kh[((size_t)(b * NKV + h) * SEQ + s) * HD + d] = (_Float16)out;
    } else {                            // V: convert only
        int h = slot - NH - NKV;
        float x = Vf[(size_t)tok * (NKV * HD) + (size_t)h * HD + d];
        Vh[((size_t)(b * NKV + h) * SEQ + s) * HD + d] = (_Float16)x;
    }
}

// ---------------------------------------------------------------------------
// Flash attention. grid (SEQ/64, NH, B), block 128 = 4 waves.
// Wave w owns 16 q-rows: online-softmax stats stay inside one wave
// (16-lane shuffles match WMMA C row layout).
// ---------------------------------------------------------------------------
__global__ __launch_bounds__(128) void flash_attn(
        const _Float16* __restrict__ Qh, const _Float16* __restrict__ Kh,
        const _Float16* __restrict__ Vh, _Float16* __restrict__ Ah) {
    const int b = blockIdx.z, h = blockIdx.y, qt = blockIdx.x;
    const int kvh  = h >> 1;                     // GQA: 2 q-heads per kv-head
    const int tid  = threadIdx.x, wv = tid >> 5;
    const int lane = tid & 31, l16 = lane & 15, hi = lane >> 4;

    __shared__ _Float16 Kt[64][HD];              // 16 KB, keys row-major
    __shared__ _Float16 Vt[HD][64 + 16];         // 20 KB, V transposed [d][key]
    __shared__ _Float16 Pw[4][16][64 + 8];       //  9 KB, per-wave P tile

    const int qrow0 = qt * 64 + wv * 16;

    // Q fragments: 16 rows x 128 (4 k-slices of 32), kept in registers
    h16x16 qf[4];
    {
        const _Float16* qp =
            Qh + ((size_t)(b * NH + h) * SEQ + qrow0 + l16) * HD;
        #pragma unroll
        for (int ks = 0; ks < 4; ++ks) {
            const _Float16* p = qp + ks * 32 + hi * 8;
            #pragma unroll
            for (int i = 0; i < 8; ++i) { qf[ks][i] = p[i]; qf[ks][8 + i] = p[16 + i]; }
        }
    }

    f32x8 o[8] = {};                             // O accum: 16 x 128
    float m_i[8], l_i[8];
    #pragma unroll
    for (int v = 0; v < 8; ++v) { m_i[v] = -3.0e38f; l_i[v] = 0.0f; }

    const int rK = tid >> 1, cK = (tid & 1) * 64;   // staging slots
    const _Float16* kvbase =
        Kh + (size_t)(b * NKV + kvh) * SEQ * HD;
    const _Float16* vbase =
        Vh + (size_t)(b * NKV + kvh) * SEQ * HD;

    for (int kt = 0; kt <= qt; ++kt) {           // causal: only tiles <= qt
        __syncthreads();
        // stage K tile [64][128]
        {
            const _Float16* g = kvbase + (size_t)(kt * 64 + rK) * HD + cK;
#if HAVE_ASYNC_COPY
            #pragma unroll
            for (int j = 0; j < 8; ++j)
                async_copy16(g + j * 8, &Kt[rK][cK + j * 8]);
            if (kt < qt) __builtin_prefetch(g + 64 * HD, 0, 1);
#else
            #pragma unroll
            for (int j = 0; j < 8; ++j)
                *(h16x8*)&Kt[rK][cK + j * 8] = *(const h16x8*)(g + j * 8);
#endif
        }
        // stage V transposed [d][key] (element scatter; sync path)
        {
            const _Float16* g = vbase + (size_t)(kt * 64 + rK) * HD + cK;
            #pragma unroll
            for (int j = 0; j < 64; ++j) Vt[cK + j][rK] = g[j];
        }
#if HAVE_ASYNC_COPY
        async_wait0();
#endif
        __syncthreads();

        // ---- scores S = Q @ K^T  (4 column tiles of 16 keys) ----
        f32x8 st[4];
        #pragma unroll
        for (int nt = 0; nt < 4; ++nt) {
            f32x8 s = {};
            #pragma unroll
            for (int ks = 0; ks < 4; ++ks) {
                h16x16 bf;
                const _Float16* kp = &Kt[nt * 16 + l16][ks * 32 + hi * 8];
                #pragma unroll
                for (int i = 0; i < 8; ++i) { bf[i] = kp[i]; bf[8 + i] = kp[16 + i]; }
                s = __builtin_amdgcn_wmma_f32_16x16x32_f16(
                    false, qf[ks], false, bf, (short)0, s, false, false);
            }
            int j = kt * 64 + nt * 16 + l16;     // global key index
            #pragma unroll
            for (int v = 0; v < 8; ++v) {
                int qi = qrow0 + hi * 8 + v;
                if (j > qi) s[v] = -3.0e38f;     // causal mask
            }
            st[nt] = s;
        }

        // ---- online softmax (per-row, within one wave half) ----
        float mnew[8], scale[8];
        #pragma unroll
        for (int v = 0; v < 8; ++v) {
            float mt = fmaxf(fmaxf(st[0][v], st[1][v]), fmaxf(st[2][v], st[3][v]));
            #pragma unroll
            for (int off = 8; off; off >>= 1) mt = fmaxf(mt, __shfl_xor(mt, off, 32));
            mnew[v]  = fmaxf(m_i[v], mt);
            scale[v] = __expf(m_i[v] - mnew[v]);
        }
        #pragma unroll
        for (int nt = 0; nt < 4; ++nt)
            #pragma unroll
            for (int v = 0; v < 8; ++v) {
                float p = __expf(st[nt][v] - mnew[v]);
                st[nt][v] = p;
                Pw[wv][hi * 8 + v][nt * 16 + l16] = (_Float16)p;   // C->A relayout
            }
        #pragma unroll
        for (int v = 0; v < 8; ++v) {
            float rs = st[0][v] + st[1][v] + st[2][v] + st[3][v];
            #pragma unroll
            for (int off = 8; off; off >>= 1) rs += __shfl_xor(rs, off, 32);
            l_i[v] = l_i[v] * scale[v] + rs;
            m_i[v] = mnew[v];
            #pragma unroll
            for (int dt = 0; dt < 8; ++dt) o[dt][v] *= scale[v];
        }

        // ---- O += P @ V  (2 key slices of 32, 8 d tiles of 16) ----
        #pragma unroll
        for (int kb = 0; kb < 2; ++kb) {
            h16x16 pf;
            const _Float16* pp = &Pw[wv][l16][kb * 32 + hi * 8];
            #pragma unroll
            for (int i = 0; i < 8; ++i) { pf[i] = pp[i]; pf[8 + i] = pp[16 + i]; }
            #pragma unroll
            for (int dt = 0; dt < 8; ++dt) {
                h16x16 vf;
                const _Float16* vp = &Vt[dt * 16 + l16][kb * 32 + hi * 8];
                #pragma unroll
                for (int i = 0; i < 8; ++i) { vf[i] = vp[i]; vf[8 + i] = vp[16 + i]; }
                o[dt] = __builtin_amdgcn_wmma_f32_16x16x32_f16(
                    false, pf, false, vf, (short)0, o[dt], false, false);
            }
        }
    }

    // finalize: O /= l, write attn as f16 token-major [B*S, NH*HD]
    #pragma unroll
    for (int v = 0; v < 8; ++v) {
        float inv = 1.0f / l_i[v];
        int row = qrow0 + hi * 8 + v;
        _Float16* out =
            Ah + ((size_t)(b * SEQ) + row) * (NH * HD) + (size_t)h * HD;
        #pragma unroll
        for (int dt = 0; dt < 8; ++dt)
            out[dt * 16 + l16] = (_Float16)(o[dt][v] * inv);
    }
}

// ---------------------------------------------------------------------------
extern "C" void kernel_launch(void* const* d_in, const int* in_sizes, int n_in,
                              void* d_out, int out_size, void* d_ws, size_t ws_size,
                              hipStream_t stream) {
    const float* hs   = (const float*)d_in[0];
    const float* cosb = (const float*)d_in[1];
    const float* sinb = (const float*)d_in[2];
    const float* Wq   = (const float*)d_in[3];
    const float* Wk   = (const float*)d_in[4];
    const float* Wv   = (const float*)d_in[5];
    const float* Wo   = (const float*)d_in[6];
    const float* qw   = (const float*)d_in[7];
    const float* kw   = (const float*)d_in[8];
    float* out = (float*)d_out;

    char* ws = (char*)d_ws;
    auto alloc = [&](size_t bytes) {
        char* p = ws; ws += (bytes + 255) & ~(size_t)255; return p;
    };
    _Float16* Xh  = (_Float16*)alloc((size_t)BS * HID * 2);
    _Float16* Wqh = (_Float16*)alloc((size_t)HID * (NH * HD) * 2);
    _Float16* Wkh = (_Float16*)alloc((size_t)HID * (NKV * HD) * 2);
    _Float16* Wvh = (_Float16*)alloc((size_t)HID * (NKV * HD) * 2);
    _Float16* Woh = (_Float16*)alloc((size_t)(NH * HD) * HID * 2);
    float*    Qf  = (float*)alloc((size_t)BS * (NH * HD) * 4);
    float*    Kf  = (float*)alloc((size_t)BS * (NKV * HD) * 4);
    float*    Vf  = (float*)alloc((size_t)BS * (NKV * HD) * 4);
    _Float16* Qh  = (_Float16*)alloc((size_t)BATCH * NH  * SEQ * HD * 2);
    _Float16* Kh  = (_Float16*)alloc((size_t)BATCH * NKV * SEQ * HD * 2);
    _Float16* Vh  = (_Float16*)alloc((size_t)BATCH * NKV * SEQ * HD * 2);
    _Float16* Ah  = (_Float16*)alloc((size_t)BS * (NH * HD) * 2);

    // 1) precision / layout prep
    {
        size_t n = (size_t)BS * HID;
        cvt_f32_to_h16<<<(unsigned)((n + 255) / 256), 256, 0, stream>>>(hs, Xh, n);
    }
    {
        size_t n = (size_t)HID * (NH * HD);
        transpose_to_h16<<<(unsigned)((n + 255) / 256), 256, 0, stream>>>(Wq, Wqh, HID, NH * HD);
        transpose_to_h16<<<(unsigned)((n + 255) / 256), 256, 0, stream>>>(Wo, Woh, NH * HD, HID);
    }
    {
        size_t n = (size_t)HID * (NKV * HD);
        transpose_to_h16<<<(unsigned)((n + 255) / 256), 256, 0, stream>>>(Wk, Wkh, HID, NKV * HD);
        transpose_to_h16<<<(unsigned)((n + 255) / 256), 256, 0, stream>>>(Wv, Wvh, HID, NKV * HD);
    }

    // 2) QKV projections
    gemm_h16_f32<<<dim3((NH * HD) / TN, BS / TM), 256, 0, stream>>>(
        Xh, Wqh, Qf, BS, NH * HD, HID);
    gemm_h16_f32<<<dim3((NKV * HD) / TN, BS / TM), 256, 0, stream>>>(
        Xh, Wkh, Kf, BS, NKV * HD, HID);
    gemm_h16_f32<<<dim3((NKV * HD) / TN, BS / TM), 256, 0, stream>>>(
        Xh, Wvh, Vf, BS, NKV * HD, HID);

    // 3) rmsnorm + rope + layout
    normrope<<<dim3(BS, NH + 2 * NKV), 128, 0, stream>>>(
        Qf, Kf, Vf, cosb, sinb, qw, kw, Qh, Kh, Vh);

    // 4) causal flash attention
    flash_attn<<<dim3(SEQ / 64, NH, BATCH), 128, 0, stream>>>(Qh, Kh, Vh, Ah);

    // 5) output projection -> f32 d_out
    gemm_h16_f32<<<dim3(HID / TN, BS / TM), 256, 0, stream>>>(
        Ah, Woh, out, BS, HID, NH * HD);
}
